// SV_GCN_28346784154174
// MI455X (gfx1250) — compile-verified, matrix operand order checked by reading, not run on hardware
//
#include <hip/hip_runtime.h>

// ---------------------------------------------------------------------------
// MI455X (gfx1250) implementation.
// Roofline: adj (400MB fp32) is streamed twice (data dependence through the
// SE gate) -> ~800MB; SE-net weights ~270MB; floor ~47us @ 23.3TB/s.
// bf16 WMMA (v_wmma_f32_16x16x32_bf16, fp32 accum) keeps the big GEMMs
// memory-bound; adj is converted fp32->bf16 in registers during staging.
// Double-buffered LDS + register staging overlaps the HBM stream with WMMA.
// ---------------------------------------------------------------------------

typedef __attribute__((ext_vector_type(16))) __bf16 bf16x16;
typedef __attribute__((ext_vector_type(8)))  float  f32x8;

__device__ __forceinline__ unsigned short f2bf(float f) {
  unsigned u = __builtin_bit_cast(unsigned, f);
  u += 0x7FFFu + ((u >> 16) & 1u);          // round-to-nearest-even
  return (unsigned short)(u >> 16);
}

union Frag {
  bf16x16 v;
  uint4   q[2];
};

constexpr int BM  = 128, BN = 128, BK = 32;
constexpr int LDA = 40;   // ushort stride (80B: 16B-aligned, padded vs 32)
constexpr int LDB = 40;

// ---- register-staged global fetch / LDS store (unguarded full tiles) ------
template<int AMODE>
__device__ __forceinline__ void fetch_A(const void* Av, int M, int K, int m0,
                                        int k0, int t, float4* aF, uint4* aU) {
  if (AMODE == 0) {
    const float* Af = (const float*)Av;
    for (int i = 0; i < 4; i++) {
      int idx = i * 256 + t;                 // float4 units
      int r = idx >> 3, c = (idx & 7) << 2;
      int row = m0 + r; if (row > M - 1) row = M - 1;
      const float* src = Af + (size_t)row * K + (k0 + c);
      aF[i] = *(const float4*)src;
      __builtin_prefetch(src + BK, 0, 0);    // global_prefetch_b8: next K tile
    }
  } else {
    const unsigned short* Ab = (const unsigned short*)Av;
    for (int i = 0; i < 2; i++) {
      int idx = i * 256 + t;                 // 8-ushort units
      int r = idx >> 2, c = (idx & 3) << 3;
      int row = m0 + r; if (row > M - 1) row = M - 1;
      aU[i] = *(const uint4*)(Ab + (size_t)row * K + (k0 + c));
    }
  }
}

template<int AMODE>
__device__ __forceinline__ void store_A(unsigned short* bufA, int t,
                                        const float4* aF, const uint4* aU) {
  if (AMODE == 0) {
    for (int i = 0; i < 4; i++) {
      int idx = i * 256 + t;
      int r = idx >> 3, c = (idx & 7) << 2;
      unsigned short* dst = &bufA[r * LDA + c];
      dst[0] = f2bf(aF[i].x); dst[1] = f2bf(aF[i].y);
      dst[2] = f2bf(aF[i].z); dst[3] = f2bf(aF[i].w);
    }
  } else {
    for (int i = 0; i < 2; i++) {
      int idx = i * 256 + t;
      int r = idx >> 2, c = (idx & 3) << 3;
      *(uint4*)&bufA[r * LDA + c] = aU[i];
    }
  }
}

template<int BMODE>
__device__ __forceinline__ void fetch_B(const void* Bv, const void* B2v, int Nn,
                                        int n0, int k0, int t,
                                        float4* bF, uint4* bU, float* bS) {
  if (BMODE == 1) {
    const unsigned short* Bb = (const unsigned short*)Bv;
    for (int i = 0; i < 2; i++) {
      int idx = i * 256 + t;                 // 8-ushort units along n
      int k = idx >> 4, nu = (idx & 15) << 3;
      bU[i] = *(const uint4*)(Bb + (size_t)(k0 + k) * Nn + (n0 + nu));
    }
  } else if (BMODE == 0) {
    const float* Bf = (const float*)Bv;
    for (int i = 0; i < 4; i++) {
      int idx = i * 256 + t;                 // float4 units along n
      int k = idx >> 5, nu = (idx & 31) << 2;
      bF[i] = *(const float4*)(Bf + (size_t)(k0 + k) * Nn + (n0 + nu));
    }
  } else {                                   // concat two fp32 [K x Nn/2]
    const float* Bf = (const float*)Bv;
    const float* B2 = (const float*)B2v;
    int Nh = Nn >> 1;
    for (int i = 0; i < 16; i++) {
      int idx = i * 256 + t;
      int k = idx >> 7, nn = idx & 127;
      bS[i] = (nn < Nh) ? Bf[(size_t)(k0 + k) * Nh + nn]
                        : B2[(size_t)(k0 + k) * Nh + (nn - Nh)];
    }
  }
}

template<int BMODE>
__device__ __forceinline__ void store_B(unsigned short* bufB, int t,
                                        const float4* bF, const uint4* bU,
                                        const float* bS) {
  if (BMODE == 1) {
    for (int i = 0; i < 2; i++) {
      int idx = i * 256 + t;
      int k = idx >> 4, nu = (idx & 15) << 3;
      unsigned short e[8];
      *(uint4*)e = bU[i];
      for (int j = 0; j < 8; j++) bufB[(nu + j) * LDB + k] = e[j];
    }
  } else if (BMODE == 0) {
    for (int i = 0; i < 4; i++) {
      int idx = i * 256 + t;
      int k = idx >> 5, nu = (idx & 31) << 2;
      bufB[(nu + 0) * LDB + k] = f2bf(bF[i].x);
      bufB[(nu + 1) * LDB + k] = f2bf(bF[i].y);
      bufB[(nu + 2) * LDB + k] = f2bf(bF[i].z);
      bufB[(nu + 3) * LDB + k] = f2bf(bF[i].w);
    }
  } else {
    for (int i = 0; i < 16; i++) {
      int idx = i * 256 + t;
      int k = idx >> 7, nn = idx & 127;
      bufB[nn * LDB + k] = f2bf(bS[i]);
    }
  }
}

// ---- guarded staging for the single K-tail tile (runs once) ---------------
template<int AMODE, int BMODE>
__device__ void stage_tail(const void* Av, const void* Bv, const void* B2v,
                           int M, int Nn, int K, int m0, int n0, int k0, int t,
                           unsigned short* bufA, unsigned short* bufB) {
  if (AMODE == 0) {
    const float* Af = (const float*)Av;
    for (int i = 0; i < 4; i++) {
      int idx = i * 256 + t;
      int r = idx >> 3, c = (idx & 7) << 2;
      int row = m0 + r; if (row > M - 1) row = M - 1;
      const float* src = Af + (size_t)row * K + (k0 + c);
      unsigned short* dst = &bufA[r * LDA + c];
      for (int j = 0; j < 4; j++)
        dst[j] = (k0 + c + j < K) ? f2bf(src[j]) : (unsigned short)0;
    }
  } else {
    const unsigned short* Ab = (const unsigned short*)Av;
    for (int i = 0; i < 2; i++) {
      int idx = i * 256 + t;
      int r = idx >> 2, c = (idx & 3) << 3;
      int row = m0 + r; if (row > M - 1) row = M - 1;
      const unsigned short* src = Ab + (size_t)row * K + (k0 + c);
      for (int j = 0; j < 8; j++)
        bufA[r * LDA + c + j] = (k0 + c + j < K) ? src[j] : (unsigned short)0;
    }
  }
  if (BMODE == 1) {
    const unsigned short* Bb = (const unsigned short*)Bv;
    for (int i = 0; i < 2; i++) {
      int idx = i * 256 + t;
      int k = idx >> 4, nu = (idx & 15) << 3;
      for (int j = 0; j < 8; j++)
        bufB[(nu + j) * LDB + k] =
            (k0 + k < K) ? Bb[(size_t)(k0 + k) * Nn + (n0 + nu + j)]
                         : (unsigned short)0;
    }
  } else if (BMODE == 0) {
    const float* Bf = (const float*)Bv;
    for (int i = 0; i < 4; i++) {
      int idx = i * 256 + t;
      int k = idx >> 5, nu = (idx & 31) << 2;
      for (int j = 0; j < 4; j++) {
        float v = (k0 + k < K) ? Bf[(size_t)(k0 + k) * Nn + (n0 + nu + j)] : 0.f;
        bufB[(nu + j) * LDB + k] = f2bf(v);
      }
    }
  } else {
    const float* Bf = (const float*)Bv;
    const float* B2 = (const float*)B2v;
    int Nh = Nn >> 1;
    for (int i = 0; i < 16; i++) {
      int idx = i * 256 + t;
      int k = idx >> 7, nn = idx & 127;
      float v = 0.f;
      if (k0 + k < K)
        v = (nn < Nh) ? Bf[(size_t)(k0 + k) * Nh + nn]
                      : B2[(size_t)(k0 + k) * Nh + (nn - Nh)];
      bufB[nn * LDB + k] = f2bf(v);
    }
  }
}

// AMODE: 0 = A fp32 row-major, 1 = A bf16 row-major
// BMODE: 0 = B fp32 row-major, 1 = B bf16 row-major, 2 = concat two fp32 [K x Nn/2]
// EPI:   0 = store bf16 acc, 1 = store bf16 relu(acc+bias[n])*scale[m], 2 = store f32 acc
// Requires: K >= BK, Nn % BN == 0 within launched grid columns.
template<int AMODE, int BMODE, int EPI>
__global__ __launch_bounds__(256, 2) void wmma_gemm(
    const void* __restrict__ Av, const void* __restrict__ Bv,
    const void* __restrict__ B2v, void* __restrict__ Cv,
    const float* __restrict__ bias, const float* __restrict__ scale,
    int M, int Nn, int K)
{
  __shared__ unsigned short lA[2][BM * LDA];   // [m][k] bf16, double-buffered
  __shared__ unsigned short lB[2][BN * LDB];   // [n][k] bf16 (transposed tile)

  const int t    = threadIdx.x;
  const int wave = t >> 5;
  const int lane = t & 31;
  const int half = lane >> 4;
  const int lm   = lane & 15;
  const int m0   = blockIdx.y * BM;
  const int n0   = blockIdx.x * BN;
  const int wm   = (wave & 3) * 32;   // wave m-offset inside block tile
  const int wn   = (wave >> 2) * 64;  // wave n-offset inside block tile

  const f32x8 zero8 = {0.f,0.f,0.f,0.f,0.f,0.f,0.f,0.f};
  f32x8 acc[2][4];
  for (int i = 0; i < 2; i++) for (int j = 0; j < 4; j++) acc[i][j] = zero8;

  const int nfull  = K / BK;
  const int rem    = K - nfull * BK;
  const int nsteps = nfull + (rem ? 1 : 0);

  float4 aF[(AMODE == 0) ? 4 : 1];
  uint4  aU[(AMODE == 1) ? 2 : 1];
  float4 bF[(BMODE == 0) ? 4 : 1];
  uint4  bU[(BMODE == 1) ? 2 : 1];
  float  bS[(BMODE == 2) ? 16 : 1];

  // prologue: stage tile 0 (always a full tile since K >= BK)
  fetch_A<AMODE>(Av, M, K, m0, 0, t, aF, aU);
  fetch_B<BMODE>(Bv, B2v, Nn, n0, 0, t, bF, bU, bS);
  store_A<AMODE>(lA[0], t, aF, aU);
  store_B<BMODE>(lB[0], t, bF, bU, bS);

  for (int s = 0; s < nsteps; s++) {
    __syncthreads();
    const int nxt = s + 1;
    if (nxt < nfull) {                       // issue next tile's global loads
      fetch_A<AMODE>(Av, M, K, m0, nxt * BK, t, aF, aU);
      fetch_B<BMODE>(Bv, B2v, Nn, n0, nxt * BK, t, bF, bU, bS);
    }

    const unsigned short* bufA = lA[s & 1];
    const unsigned short* bufB = lB[s & 1];

    // fragments: two 16B ds_load_b128 per frag per lane
    Frag fa[2], fb[4];
    for (int mt = 0; mt < 2; mt++) {
      const unsigned short* p = &bufA[(wm + mt * 16 + lm) * LDA + 8 * half];
      fa[mt].q[0] = *(const uint4*)p;        // k = 8h .. 8h+7
      fa[mt].q[1] = *(const uint4*)(p + 16); // k = 16+8h .. 16+8h+7
    }
    for (int nt = 0; nt < 4; nt++) {
      const unsigned short* p = &bufB[(wn + nt * 16 + lm) * LDB + 8 * half];
      fb[nt].q[0] = *(const uint4*)p;
      fb[nt].q[1] = *(const uint4*)(p + 16);
    }
    for (int mt = 0; mt < 2; mt++)
      for (int nt = 0; nt < 4; nt++)
        acc[mt][nt] = __builtin_amdgcn_wmma_f32_16x16x32_bf16(
            false, fa[mt].v, false, fb[nt].v, (short)0, acc[mt][nt],
            false, false);

    if (nxt < nfull) {                       // convert + park into other buffer
      store_A<AMODE>(lA[nxt & 1], t, aF, aU);
      store_B<BMODE>(lB[nxt & 1], t, bF, bU, bS);
    } else if (nxt == nfull && rem) {        // single guarded K-tail tile
      stage_tail<AMODE, BMODE>(Av, Bv, B2v, M, Nn, K, m0, n0, nxt * BK, t,
                               lA[nxt & 1], lB[nxt & 1]);
    }
  }

  // ---- fused epilogue ----
  for (int mt = 0; mt < 2; mt++) {
    for (int nt = 0; nt < 4; nt++) {
      int n = n0 + wn + nt * 16 + lm;
      if (n >= Nn) continue;
      for (int v = 0; v < 8; v++) {
        int m = m0 + wm + mt * 16 + 8 * half + v;
        if (m >= M) continue;
        float r = acc[mt][nt][v];
        if (EPI == 1) r = fmaxf(r + bias[n], 0.f) * scale[m];
        if (EPI == 2) ((float*)Cv)[(size_t)m * Nn + n] = r;
        else          ((unsigned short*)Cv)[(size_t)m * Nn + n] = f2bf(r);
      }
    }
  }
}

// ---------------------------------------------------------------------------
// SE-net path: row-sum, two LDS-cached GEMVs (weight-streaming bound).
// ---------------------------------------------------------------------------
__global__ void rowsum_kernel(const float* __restrict__ x, float* __restrict__ s_in,
                              int F) {
  const int row = blockIdx.x;
  const int t = threadIdx.x;
  const float* r = x + (size_t)row * F;
  float s = 0.f;
  for (int j = t; j < F; j += 128) s += r[j];
  for (int off = 16; off; off >>= 1) s += __shfl_down(s, off, 32);
  __shared__ float ws[4];
  if ((t & 31) == 0) ws[t >> 5] = s;
  __syncthreads();
  if (t == 0) s_in[row] = ws[0] + ws[1] + ws[2] + ws[3];
}

template<int ACT>  // 0 = relu, 1 = sigmoid
__global__ void gemv_kernel(const float* __restrict__ vec, const float* __restrict__ W,
                            const float* __restrict__ b, float* __restrict__ out,
                            int K, int N) {
  __shared__ float vc[256];
  const int n = blockIdx.x * 256 + threadIdx.x;
  float acc = 0.f;
  for (int k0 = 0; k0 < K; k0 += 256) {
    __syncthreads();
    int kk = k0 + threadIdx.x;
    vc[threadIdx.x] = (kk < K) ? vec[kk] : 0.f;
    __syncthreads();
    const int kend = (K - k0 < 256) ? (K - k0) : 256;
    if (n < N) {
      const float* Wp = W + (size_t)k0 * N + n;   // coalesced along n
      for (int k = 0; k < kend; k++) acc += vc[k] * Wp[(size_t)k * N];
    }
  }
  if (n < N) {
    acc += b[n];
    out[n] = (ACT == 0) ? fmaxf(acc, 0.f) : 1.f / (1.f + __expf(-acc));
  }
}

// z = eps*exp(logstd)+mean, log_softmax over 64 classes; one wave32 per row.
__global__ void final_kernel(const float* __restrict__ ml, const float* __restrict__ eps,
                             const float* __restrict__ b_mean, const float* __restrict__ b_ls,
                             float* __restrict__ out, int M, int C /* must be 64 */) {
  const int row  = blockIdx.x * 8 + (threadIdx.x >> 5);
  const int lane = threadIdx.x & 31;
  if (row >= M) return;
  const float* r = ml + (size_t)row * (2 * C);
  float z[2];
  for (int i = 0; i < 2; i++) {
    int j = lane + i * 32;
    float mean = r[j] + b_mean[j];
    float ls   = r[C + j] + b_ls[j];
    z[i] = eps[(size_t)row * C + j] * __expf(ls) + mean;
  }
  float mx = fmaxf(z[0], z[1]);
  for (int off = 16; off; off >>= 1) mx = fmaxf(mx, __shfl_xor(mx, off, 32));
  float s = __expf(z[0] - mx) + __expf(z[1] - mx);
  for (int off = 16; off; off >>= 1) s += __shfl_xor(s, off, 32);
  float lse = mx + __logf(s);
  out[(size_t)row * C + lane]      = z[0] - lse;
  out[(size_t)row * C + lane + 32] = z[1] - lse;
}

// ---------------------------------------------------------------------------
extern "C" void kernel_launch(void* const* d_in, const int* in_sizes, int n_in,
                              void* d_out, int out_size, void* d_ws, size_t ws_size,
                              hipStream_t stream) {
  const float* x     = (const float*)d_in[0];
  const float* adj   = (const float*)d_in[1];
  const float* eps   = (const float*)d_in[2];
  const float* w_gc1 = (const float*)d_in[3];
  const float* b_gc1 = (const float*)d_in[4];
  const float* w11   = (const float*)d_in[5];
  const float* b11   = (const float*)d_in[6];
  const float* w12   = (const float*)d_in[7];
  const float* b12   = (const float*)d_in[8];
  const float* sw1   = (const float*)d_in[9];
  const float* sb1   = (const float*)d_in[10];
  const float* sw2   = (const float*)d_in[11];
  const float* sb2   = (const float*)d_in[12];

  const int N  = in_sizes[12];      // 10000 nodes
  const int H  = in_sizes[4];       // 256 hidden
  const int C  = in_sizes[6];       // 64 classes
  const int F  = in_sizes[0] / N;   // 512 features
  const int SH = in_sizes[10];      // 3333 SE hidden
  const int C2 = 2 * C;             // 128 (dual heads fused)

  // workspace carve-out (~18 MB total)
  char* wbase = (char*)d_ws;
  size_t off = 0;
  auto carve = [&](size_t bytes) -> void* {
    void* p = wbase + off;
    off = (off + bytes + 255) & ~(size_t)255;
    return p;
  };
  unsigned short* xw    = (unsigned short*)carve((size_t)N * H  * 2);
  unsigned short* gated = (unsigned short*)carve((size_t)N * H  * 2);
  unsigned short* gw    = (unsigned short*)carve((size_t)N * C2 * 2);
  float* ml    = (float*)carve((size_t)N * C2 * 4);
  float* s_in  = (float*)carve((size_t)N  * 4);
  float* s_hid = (float*)carve((size_t)SH * 4);
  float* score = (float*)carve((size_t)N  * 4);
  (void)ws_size; (void)n_in; (void)out_size;

  // SE-net gate (depends only on x) first, so GEMM1 can fuse the gating.
  rowsum_kernel<<<N, 128, 0, stream>>>(x, s_in, F);
  gemv_kernel<0><<<(SH + 255) / 256, 256, 0, stream>>>(s_in, sw1, sb1, s_hid, N, SH);
  gemv_kernel<1><<<(N  + 255) / 256, 256, 0, stream>>>(s_hid, sw2, sb2, score, SH, N);

  const dim3 blk(256);
  const int  gy = (N + BM - 1) / BM;       // 79 M-blocks

  // xw = x @ w_gc1                                  (fp32 x, fp32 W -> bf16)
  wmma_gemm<0, 0, 0><<<dim3((H + BN - 1) / BN, gy), blk, 0, stream>>>(
      x, w_gc1, nullptr, xw, nullptr, nullptr, N, H, F);
  // gated = relu(adj @ xw + b_gc1) * score          (fused epilogue -> bf16)
  wmma_gemm<0, 1, 1><<<dim3((H + BN - 1) / BN, gy), blk, 0, stream>>>(
      adj, xw, nullptr, gated, b_gc1, score, N, H, N);
  // gw = gated @ [w_fc11 | w_fc12]                  (bf16 A, concat fp32 B -> bf16)
  wmma_gemm<1, 2, 0><<<dim3((C2 + BN - 1) / BN, gy), blk, 0, stream>>>(
      gated, w11, w12, gw, nullptr, nullptr, N, C2, H);
  // ml = adj @ gw                                   (fp32 out: [mean | logstd] pre-bias)
  wmma_gemm<0, 1, 2><<<dim3((C2 + BN - 1) / BN, gy), blk, 0, stream>>>(
      adj, gw, nullptr, ml, nullptr, nullptr, N, C2, N);
  // z = eps*exp(logstd)+mean ; log_softmax
  final_kernel<<<(N + 7) / 8, 256, 0, stream>>>(ml, eps, b11, b12, (float*)d_out, N, C);
}
